// CNF_plain_23278722744619
// MI455X (gfx1250) — matrix operand order repaired and checked by reading, not compiled, not run on hardware
//
#include <hip/hip_runtime.h>

typedef __attribute__((ext_vector_type(16))) _Float16 v16h;
typedef __attribute__((ext_vector_type(8)))  _Float16 v8h;
typedef __attribute__((ext_vector_type(8)))  float    v8f;

#define NWAVE 2              // waves per block
#define MT 2                 // 16-row M-tiles per wave
#define SPW (16 * MT)        // samples per wave = 32
#define RS 136               // padded LDS row stride in halfs (272B = 17*16B)

// Branch-free tanh: hardware V_TANH_F32 if available, else exp2/rcp formula.
__device__ __forceinline__ float fast_tanh(float x) {
#if __has_builtin(__builtin_amdgcn_tanhf)
  return __builtin_amdgcn_tanhf(x);
#else
  float e = __builtin_amdgcn_exp2f(__builtin_fabsf(x) * 2.8853900817779268f);
  float r = 1.0f - 2.0f * __builtin_amdgcn_rcpf(e + 1.0f);
  return __builtin_copysignf(r, x);
#endif
}

// ---------------------------------------------------------------------------
// Prep: f16 transposed weights in workspace.
//   W2T[c*128+k] = W2[k,c]
//   Mc [c*128+k] = W2[k,c] * G[c,k],  G[c,k] = sum_d W3[c,d]*W1[d,k]
//   W3T[n*128+k] = (n<8) ? W3[k,n] : 0   (padded to 16 columns)
// ---------------------------------------------------------------------------
__global__ __launch_bounds__(256) void cnf_prep(
    const float* __restrict__ W1, const float* __restrict__ W2,
    const float* __restrict__ W3,
    _Float16* __restrict__ W2T, _Float16* __restrict__ Mc,
    _Float16* __restrict__ W3T) {
  int gid = blockIdx.x * 256 + threadIdx.x;   // 0..16383
  int c = gid >> 7, k = gid & 127;
  float g = 0.f;
#pragma unroll
  for (int d = 0; d < 8; ++d) g += W3[c * 8 + d] * W1[d * 128 + k];
  float w2 = W2[k * 128 + c];
  W2T[c * 128 + k] = (_Float16)w2;
  Mc[c * 128 + k]  = (_Float16)(w2 * g);
  if (gid < 16 * 128) {
    int n = gid >> 7;
    W3T[gid] = (n < 8) ? (_Float16)W3[k * 8 + n] : (_Float16)0.f;
  }
}

// ---------------------------------------------------------------------------
// Main: each wave owns 32 samples (two 16-row M-tiles; B-tiles feed 2 WMMAs).
// ---------------------------------------------------------------------------
__global__ __launch_bounds__(NWAVE * 32) void cnf_main(
    const float* __restrict__ z, const float* __restrict__ t,
    const float* __restrict__ W1, const float* __restrict__ b1,
    const float* __restrict__ b2, const float* __restrict__ b3,
    const _Float16* __restrict__ W2T, const _Float16* __restrict__ Mc,
    const _Float16* __restrict__ W3T,
    float* __restrict__ out_dz, float* __restrict__ out_dlogp) {

  __shared__ __attribute__((aligned(16))) _Float16 h1s[NWAVE * SPW * RS];
  __shared__ __attribute__((aligned(16))) _Float16 s1s[NWAVE * SPW * RS];
  __shared__ __attribute__((aligned(16))) _Float16 h2s[NWAVE * SPW * RS];
  __shared__ float zs[NWAVE * SPW * 8];
  __shared__ float trbuf[NWAVE * 32 * 16];
  __shared__ float W1s[9 * 128];
  __shared__ float b1s[128], b2s[128], b3s[8];

  const int tid  = threadIdx.x;
  const int w    = tid >> 5;
  const int lane = tid & 31;
  const long sampleBase = ((long)blockIdx.x * NWAVE + w) * SPW;

  for (int i = tid; i < 9 * 128; i += NWAVE * 32) W1s[i] = W1[i];
  for (int i = tid; i < 128; i += NWAVE * 32) { b1s[i] = b1[i]; b2s[i] = b2[i]; }
  if (tid < 8) b3s[tid] = b3[tid];
#pragma unroll
  for (int j = 0; j < SPW / 4; ++j)                       // 32*8 floats per wave
    zs[w * SPW * 8 + lane + 32 * j] = z[sampleBase * 8 + lane + 32 * j];
  const float tval = t[0];
  __syncthreads();

  _Float16* h1w = &h1s[w * SPW * RS];
  _Float16* s1w = &s1s[w * SPW * RS];
  _Float16* h2w = &h2s[w * SPW * RS];

  // ---- layer 1: each lane owns 4 hidden columns for all 32 samples
  {
    const int c0 = lane * 4;
    float wcol[4][9], bcol[4];
#pragma unroll
    for (int cc = 0; cc < 4; ++cc) {
      bcol[cc] = b1s[c0 + cc];
#pragma unroll
      for (int k = 0; k < 9; ++k) wcol[cc][k] = W1s[k * 128 + c0 + cc];
    }
    for (int b = 0; b < SPW; ++b) {
      float zr[8];
#pragma unroll
      for (int k = 0; k < 8; ++k) zr[k] = zs[w * SPW * 8 + b * 8 + k];
#pragma unroll
      for (int cc = 0; cc < 4; ++cc) {
        float acc = bcol[cc] + tval * wcol[cc][8];
#pragma unroll
        for (int k = 0; k < 8; ++k) acc += zr[k] * wcol[cc][k];
        float th = fast_tanh(acc);
        h1w[b * RS + c0 + cc] = (_Float16)th;
        s1w[b * RS + c0 + cc] = (_Float16)(1.f - th * th);
      }
    }
  }
  __syncthreads();

  const int m  = lane & 15;    // A-row within tile / B-column within tile
  const int hi = lane >> 4;    // lane half -> K sub-range
  const int kb = hi * 8;

  // ---- preload all A-tiles into registers (invariant across nt)
  // A layout (16x32 f16): halfs 0-7 = K kb..kb+7, halfs 8-15 = K kb+16..kb+23
  v16h a1t[MT * 4], a2t[MT * 4];
#pragma unroll
  for (int mt = 0; mt < MT; ++mt)
#pragma unroll
    for (int kc = 0; kc < 4; ++kc) {
      const int row = mt * 16 + m, kOff = kc * 32, idx = mt * 4 + kc;
      union { v16h v; v8h h[2]; } u1, u2;
      u1.h[0] = *(const v8h*)&h1w[row * RS + kOff + kb];
      u1.h[1] = *(const v8h*)&h1w[row * RS + kOff + 16 + kb];
      u2.h[0] = *(const v8h*)&s1w[row * RS + kOff + kb];
      u2.h[1] = *(const v8h*)&s1w[row * RS + kOff + 16 + kb];
      a1t[idx] = u1.v;  a2t[idx] = u2.v;
    }

  float trPart[MT * 8];
#pragma unroll
  for (int i = 0; i < MT * 8; ++i) trPart[i] = 0.f;

  // ---- fused GEMMs: u2 = h1@W2 and v2 = s1@M, tile-by-tile over N
  for (int nt = 0; nt < 8; ++nt) {
    v8f acc2[MT], accT[MT];
#pragma unroll
    for (int mt = 0; mt < MT; ++mt) { acc2[mt] = (v8f){}; accT[mt] = (v8f){}; }
    const int col = nt * 16 + m;
#pragma unroll
    for (int kc = 0; kc < 4; ++kc) {
      const int kOff = kc * 32;
      // B layout (32x16 f16): halfs 0-7 = K kb..kb+7, halfs 8-15 = K kb+16..kb+23 of column `col`
      union { v16h v; v8h h[2]; } bW, bM;
      const _Float16* bw = &W2T[col * 128 + kOff + kb];
      const _Float16* bm = &Mc [col * 128 + kOff + kb];
      bW.h[0] = *(const v8h*)bw;  bW.h[1] = *(const v8h*)(bw + 16);
      bM.h[0] = *(const v8h*)bm;  bM.h[1] = *(const v8h*)(bm + 16);
#pragma unroll
      for (int mt = 0; mt < MT; ++mt) {
        acc2[mt] = __builtin_amdgcn_wmma_f32_16x16x32_f16(
            false, a1t[mt * 4 + kc], false, bW.v, (short)0, acc2[mt], false, false);
        accT[mt] = __builtin_amdgcn_wmma_f32_16x16x32_f16(
            false, a2t[mt * 4 + kc], false, bM.v, (short)0, accT[mt], false, false);
      }
    }
    // C layout: vgpr v -> row mt*16 + v + 8*hi (= sample), col = nt*16 + m
    const float bb2 = b2s[col];
#pragma unroll
    for (int mt = 0; mt < MT; ++mt)
#pragma unroll
      for (int v = 0; v < 8; ++v) {
        float th  = fast_tanh(acc2[mt][v] + bb2);
        float s2v = 1.f - th * th;
        h2w[(mt * 16 + v + 8 * hi) * RS + col] = (_Float16)th;
        trPart[mt * 8 + v] += accT[mt][v] * s2v;
      }
  }
  __syncthreads();

  // ---- dz = h2 @ W3 (+ b3), W3 padded to 16 columns
  {
    v8f accD[MT];
#pragma unroll
    for (int mt = 0; mt < MT; ++mt) accD[mt] = (v8f){};
#pragma unroll
    for (int kc = 0; kc < 4; ++kc) {
      const int kOff = kc * 32;
      union { v16h v; v8h h[2]; } bC;
      const _Float16* b3p = &W3T[m * 128 + kOff + kb];
      bC.h[0] = *(const v8h*)b3p;  bC.h[1] = *(const v8h*)(b3p + 16);
#pragma unroll
      for (int mt = 0; mt < MT; ++mt) {
        union { v16h v; v8h h[2]; } aH;
        aH.h[0] = *(const v8h*)&h2w[(mt * 16 + m) * RS + kOff + kb];
        aH.h[1] = *(const v8h*)&h2w[(mt * 16 + m) * RS + kOff + 16 + kb];
        accD[mt] = __builtin_amdgcn_wmma_f32_16x16x32_f16(
            false, aH.v, false, bC.v, (short)0, accD[mt], false, false);
      }
    }
    if (m < 8) {
#pragma unroll
      for (int mt = 0; mt < MT; ++mt)
#pragma unroll
        for (int v = 0; v < 8; ++v) {
          int row = mt * 16 + v + 8 * hi;
          out_dz[(sampleBase + row) * 8 + m] = accD[mt][v] + b3s[m];
        }
    }
  }

  // ---- trace reduction: sample b gets contributions from 16 lanes
#pragma unroll
  for (int i = 0; i < MT * 8; ++i) trbuf[(w * 32 + lane) * 16 + i] = trPart[i];
  __syncthreads();
  {
    const int b     = lane;              // local sample 0..31
    const int srcHi = (b >> 3) & 1;
    const int mt    = b >> 4;
    const int v     = b & 7;
    float s = 0.f;
#pragma unroll
    for (int l = 0; l < 16; ++l)
      s += trbuf[(w * 32 + srcHi * 16 + l) * 16 + mt * 8 + v];
    out_dlogp[sampleBase + b] = -s;
  }
}

extern "C" void kernel_launch(void* const* d_in, const int* in_sizes, int n_in,
                              void* d_out, int out_size, void* d_ws, size_t ws_size,
                              hipStream_t stream) {
  (void)n_in; (void)out_size; (void)ws_size;
  const float* z  = (const float*)d_in[0];
  // d_in[1] = logp_z (unused by the outputs)
  const float* t  = (const float*)d_in[2];
  const float* W1 = (const float*)d_in[3];
  const float* b1 = (const float*)d_in[4];
  const float* W2 = (const float*)d_in[5];
  const float* b2 = (const float*)d_in[6];
  const float* W3 = (const float*)d_in[7];
  const float* b3 = (const float*)d_in[8];

  const long B = (long)in_sizes[0] / 8;   // 131072

  _Float16* W2T = (_Float16*)d_ws;                       // 32 KB
  _Float16* Mc  = (_Float16*)((char*)d_ws + 32768);      // 32 KB
  _Float16* W3T = (_Float16*)((char*)d_ws + 65536);      //  4 KB

  float* out_dz    = (float*)d_out;
  float* out_dlogp = out_dz + B * 8;

  cnf_prep<<<64, 256, 0, stream>>>(W1, W2, W3, W2T, Mc, W3T);
  cnf_main<<<(int)(B / (SPW * NWAVE)), NWAVE * 32, 0, stream>>>(
      z, t, W1, b1, b2, b3, W2T, Mc, W3T, out_dz, out_dlogp);
}
